// PointerGenerator_13065290514977
// MI455X (gfx1250) — matrix operand order repaired and checked by reading, not compiled, not run on hardware
//
#include <hip/hip_runtime.h>

typedef __attribute__((ext_vector_type(16))) __bf16 v16bf;
typedef __attribute__((ext_vector_type(8)))  __bf16 v8bf;
typedef __attribute__((ext_vector_type(8)))  float  v8f;

constexpr int NB = 32;     // batches
constexpr int LD = 512;    // decoder length (rows of scores)
constexpr int LE = 2048;   // encoder length (cols of scores, softmax axis)
constexpr int DK = 512;    // feature dim (K)
constexpr int WAVES = 16;  // 512 threads per block
constexpr int JT = 8;      // j-tiles (16 cols each) per wave -> 128 cols/wave

constexpr size_t DEC_N = (size_t)NB * LD * DK;   // 8,388,608
constexpr size_t ENC_N = (size_t)NB * LE * DK;   // 33,554,432
constexpr size_t WS_NEED = (2 * DEC_N + 2 * ENC_N) * sizeof(__bf16); // ~168 MB

__device__ __forceinline__ v16bf cat16(v8bf a, v8bf b) {
  return __builtin_shufflevector(a, b, 0, 1, 2, 3, 4, 5, 6, 7,
                                       8, 9, 10, 11, 12, 13, 14, 15);
}

__device__ __forceinline__ void split_bf16(const float (&x)[16], v16bf& hi, v16bf& lo) {
#pragma unroll
  for (int e = 0; e < 16; ++e) {
    float v = x[e];
    __bf16 h = (__bf16)v;
    hi[e] = h;
    lo[e] = (__bf16)(v - (float)h);
  }
}

// ---- one-time f32 -> (bf16 hi, bf16 lo) split, 8 elements/thread ----
__global__ __launch_bounds__(256)
void split_kernel(const float* __restrict__ in, __bf16* __restrict__ hi,
                  __bf16* __restrict__ lo, unsigned n8) {
  unsigned g = blockIdx.x * blockDim.x + threadIdx.x;
  if (g >= n8) return;
  size_t base = (size_t)g * 8;
  float4 x0 = *(const float4*)(in + base);
  float4 x1 = *(const float4*)(in + base + 4);
  float x[8] = {x0.x, x0.y, x0.z, x0.w, x1.x, x1.y, x1.z, x1.w};
  v8bf h, l;
#pragma unroll
  for (int e = 0; e < 8; ++e) {
    __bf16 hh = (__bf16)x[e];
    h[e] = hh;
    l[e] = (__bf16)(x[e] - (float)hh);
  }
  *(v8bf*)(hi + base) = h;
  *(v8bf*)(lo + base) = l;
}

// PRE=true: load pre-split bf16 hi/lo fragments (no VALU in the K loop).
// PRE=false: fused f32->bf16 split (fallback when d_ws is too small).
template <bool PRE>
__global__ __launch_bounds__(WAVES * 32, 1)
void xattn_softmax_kernel(const float* __restrict__ dec,
                          const float* __restrict__ enc,
                          const __bf16* __restrict__ dhi,
                          const __bf16* __restrict__ dlo,
                          const __bf16* __restrict__ ehi,
                          const __bf16* __restrict__ elo,
                          float* __restrict__ out) {
  __shared__ float red_max[WAVES][16];
  __shared__ float red_sum[WAVES][16];

  const int blk   = blockIdx.x;
  const int b     = blk >> 5;        // / (LD/16)
  const int itile = blk & 31;
  const int i0    = itile * 16;

  const int tid  = threadIdx.x;
  const int wave = tid >> 5;
  const int lane = tid & 31;
  const int lm   = lane & 15;
  const int half = lane >> 4;
  const int jb   = wave * (JT * 16);

  // A fragment: lane holds row M = lane&15; K window depends on half.
  const size_t arow = ((size_t)b * LD + i0 + lm) * DK + half * 16;
  // B fragment: lane holds col N = lane&15; 16 contiguous K at half*16.
  const size_t brow = ((size_t)b * LE + jb + lm) * DK + half * 16;

  v8f acc[JT];
#pragma unroll
  for (int t = 0; t < JT; ++t) {
    v8f z = {0.f, 0.f, 0.f, 0.f, 0.f, 0.f, 0.f, 0.f};
    acc[t] = z;
  }

#pragma unroll 1
  for (int k0 = 0; k0 < DK; k0 += 32) {
    v16bf ah, al;
    if constexpr (PRE) {
      const __bf16* Ah = dhi + arow;
      const __bf16* Al = dlo + arow;
      ah = cat16(*(const v8bf*)(Ah + k0), *(const v8bf*)(Ah + k0 + 8));
      al = cat16(*(const v8bf*)(Al + k0), *(const v8bf*)(Al + k0 + 8));
    } else {
      const float* Ap = dec + arow;
      float4 a0 = *(const float4*)(Ap + k0);
      float4 a1 = *(const float4*)(Ap + k0 + 4);
      float4 a2 = *(const float4*)(Ap + k0 + 8);
      float4 a3 = *(const float4*)(Ap + k0 + 12);
      float a32[16] = {a0.x, a0.y, a0.z, a0.w, a1.x, a1.y, a1.z, a1.w,
                       a2.x, a2.y, a2.z, a2.w, a3.x, a3.y, a3.z, a3.w};
      split_bf16(a32, ah, al);
    }

#pragma unroll
    for (int t = 0; t < JT; ++t) {
      v16bf bh, bl;
      if constexpr (PRE) {
        const __bf16* Bh = ehi + brow + (size_t)t * 16 * DK + k0;
        const __bf16* Bl = elo + brow + (size_t)t * 16 * DK + k0;
        bh = cat16(*(const v8bf*)Bh, *(const v8bf*)(Bh + 8));
        bl = cat16(*(const v8bf*)Bl, *(const v8bf*)(Bl + 8));
      } else {
        const float* Bp = enc + brow + (size_t)t * 16 * DK + k0;
        float4 b0 = *(const float4*)(Bp);
        float4 b1 = *(const float4*)(Bp + 4);
        float4 b2 = *(const float4*)(Bp + 8);
        float4 b3 = *(const float4*)(Bp + 12);
        float b32[16] = {b0.x, b0.y, b0.z, b0.w, b1.x, b1.y, b1.z, b1.w,
                         b2.x, b2.y, b2.z, b2.w, b3.x, b3.y, b3.z, b3.w};
        split_bf16(b32, bh, bl);
      }
      acc[t] = __builtin_amdgcn_wmma_f32_16x16x32_bf16(
          false, ah, false, bh, (short)0, acc[t], false, false);
      acc[t] = __builtin_amdgcn_wmma_f32_16x16x32_bf16(
          false, ah, false, bl, (short)0, acc[t], false, false);
      acc[t] = __builtin_amdgcn_wmma_f32_16x16x32_bf16(
          false, al, false, bh, (short)0, acc[t], false, false);
    }
  }

  // C/D layout: lane L, VGPR r -> row m = half*8 + r, col n = jb + t*16 + (L&15)

  // ---- row max: tiles -> 16-lane half -> cross-wave via LDS ----
  float rmax[8];
#pragma unroll
  for (int r = 0; r < 8; ++r) {
    float m = acc[0][r];
#pragma unroll
    for (int t = 1; t < JT; ++t) m = fmaxf(m, acc[t][r]);
    m = fmaxf(m, __shfl_xor(m, 1, 32));
    m = fmaxf(m, __shfl_xor(m, 2, 32));
    m = fmaxf(m, __shfl_xor(m, 4, 32));
    m = fmaxf(m, __shfl_xor(m, 8, 32));
    rmax[r] = m;
  }
  if (lm == 0) {
#pragma unroll
    for (int r = 0; r < 8; ++r) red_max[wave][half * 8 + r] = rmax[r];
  }
  __syncthreads();

  float gmax[8];
#pragma unroll
  for (int r = 0; r < 8; ++r) {
    float m = red_max[0][half * 8 + r];
#pragma unroll
    for (int w = 1; w < WAVES; ++w) m = fmaxf(m, red_max[w][half * 8 + r]);
    gmax[r] = m;
  }

  // ---- exp + row sum ----
  float rsum[8];
#pragma unroll
  for (int r = 0; r < 8; ++r) {
    float s = 0.f;
#pragma unroll
    for (int t = 0; t < JT; ++t) {
      float e = __expf(acc[t][r] - gmax[r]);
      acc[t][r] = e;
      s += e;
    }
    s += __shfl_xor(s, 1, 32);
    s += __shfl_xor(s, 2, 32);
    s += __shfl_xor(s, 4, 32);
    s += __shfl_xor(s, 8, 32);
    rsum[r] = s;
  }
  if (lm == 0) {
#pragma unroll
    for (int r = 0; r < 8; ++r) red_sum[wave][half * 8 + r] = rsum[r];
  }
  __syncthreads();

  float rinv[8];
#pragma unroll
  for (int r = 0; r < 8; ++r) {
    float s = red_sum[0][half * 8 + r];
#pragma unroll
    for (int w = 1; w < WAVES; ++w) s += red_sum[w][half * 8 + r];
    rinv[r] = 1.0f / s;
  }

  // ---- normalize + store (lanes 0..15 / 16..31 each cover 64B-contiguous runs) ----
  float* Orow = out + ((size_t)b * LD + i0) * LE;
#pragma unroll
  for (int t = 0; t < JT; ++t) {
    const int col = jb + t * 16 + lm;
#pragma unroll
    for (int r = 0; r < 8; ++r) {
      const int row = half * 8 + r;
      Orow[(size_t)row * LE + col] = acc[t][r] * rinv[r];
    }
  }
}

extern "C" void kernel_launch(void* const* d_in, const int* in_sizes, int n_in,
                              void* d_out, int out_size, void* d_ws, size_t ws_size,
                              hipStream_t stream) {
  (void)in_sizes; (void)n_in; (void)out_size;
  const float* dec = (const float*)d_in[0];   // (32, 512, 512)  f32
  const float* enc = (const float*)d_in[1];   // (32, 2048, 512) f32
  float* out = (float*)d_out;                 // (32, 512, 2048) f32

  dim3 grid(NB * (LD / 16));   // 1024 workgroups: one per (batch, 16-row block)
  dim3 block(WAVES * 32);      // 512 threads = 16 waves

  if (ws_size >= WS_NEED) {
    __bf16* dhi = (__bf16*)d_ws;
    __bf16* dlo = dhi + DEC_N;
    __bf16* ehi = dlo + DEC_N;
    __bf16* elo = ehi + ENC_N;
    const unsigned dn8 = (unsigned)(DEC_N / 8);
    const unsigned en8 = (unsigned)(ENC_N / 8);
    split_kernel<<<(dn8 + 255) / 256, 256, 0, stream>>>(dec, dhi, dlo, dn8);
    split_kernel<<<(en8 + 255) / 256, 256, 0, stream>>>(enc, ehi, elo, en8);
    xattn_softmax_kernel<true><<<grid, block, 0, stream>>>(
        dec, enc, dhi, dlo, ehi, elo, out);
  } else {
    xattn_softmax_kernel<false><<<grid, block, 0, stream>>>(
        dec, enc, nullptr, nullptr, nullptr, nullptr, out);
  }
}
